// TrnModel_50508815401074
// MI455X (gfx1250) — matrix operand order repaired and checked by reading, not compiled, not run on hardware
//
#include <hip/hip_runtime.h>
#include <hip/hip_bf16.h>

// ---------------------------------------------------------------------------
// Problem constants (from the reference)
// ---------------------------------------------------------------------------
#define Hh      256          // hidden size H
#define EIN     512          // H + 2*AE + 2*ND = 256+128+128
#define ATTH    128
#define NMAXX   1024
#define NNODES  8192         // B*C*NMAX
#define MEDGES  250000
#define POUT    250000

typedef __attribute__((ext_vector_type(16))) __bf16 v16bf;
typedef __attribute__((ext_vector_type(8)))  __bf16 v8bf;
typedef __attribute__((ext_vector_type(8)))  float  v8f;

union V16U { v16bf v; v8bf h[2]; };

// A fragment (16x32 bf16, MxK) from an LDS tile, row-major with `stride` elems.
// Contiguous-k order (same as frag_b): lane L holds row m=L&15; g=L>>4; elems j = K[k0+16g+j].
// -> two ds_load_b128 per lane.
__device__ __forceinline__ v16bf frag_a(const __bf16* tile, int stride, int k0, int lane) {
    const int m = lane & 15, g = lane >> 4;
    V16U u;
    u.h[0] = *(const v8bf*)(tile + m * stride + k0 + 16 * g);
    u.h[1] = *(const v8bf*)(tile + m * stride + k0 + 16 * g + 8);
    return u.v;
}

// B fragment (32x16 bf16, KxN) from a global weight stored N-major (out x in,
// i.e. W[n][k], which is exactly how the reference weights are laid out).
// Contiguous-k order (same as frag_a): lane L holds col n=L&15; g=L>>4; elems j = K[k0+16g+j].
// -> two global_load_b128 per lane.
__device__ __forceinline__ v16bf frag_b(const __bf16* __restrict__ W, int Kdim,
                                        int n0, int k0, int lane) {
    const int n = lane & 15, g = lane >> 4;
    const __bf16* p = W + (size_t)(n0 + n) * Kdim + k0 + 16 * g;
    V16U u;
    u.h[0] = *(const v8bf*)(p);
    u.h[1] = *(const v8bf*)(p + 8);
    return u.v;
}

__device__ __forceinline__ v8f wmma_bf16(v16bf a, v16bf b, v8f c) {
    // 8-arg form: (neg_a, A, neg_b, B, c_mod, C, reuse_a, reuse_b)
    return __builtin_amdgcn_wmma_f32_16x16x32_bf16(false, a, false, b,
                                                   (short)0, c, false, false);
}

__device__ __forceinline__ float sigmoidf(float x) { return 1.0f / (1.0f + expf(-x)); }

// Hardware fp32 atomic add (segment_sum target is L2-resident -> native RMW).
__device__ __forceinline__ void atomic_add_f32(float* p, float v) {
    asm volatile("global_atomic_add_f32 %0, %1, off"
                 :: "v"((unsigned long long)p), "v"(v)
                 : "memory");
}

// ---------------------------------------------------------------------------
// weight f32 -> bf16
// ---------------------------------------------------------------------------
__global__ void k_f32_to_bf16(const float* __restrict__ src, __bf16* __restrict__ dst, int n) {
    for (int i = blockIdx.x * blockDim.x + threadIdx.x; i < n; i += gridDim.x * blockDim.x)
        dst[i] = (__bf16)src[i];
}

// ---------------------------------------------------------------------------
// decoder:  out(8192x256) = A2(8192x1024) @ W.T + b     (launched twice)
// block = 256 thr (8 waves), 16 rows per block; each wave owns 2 N-tiles.
// ---------------------------------------------------------------------------
__global__ void __launch_bounds__(256) k_dec_gemm(const float* __restrict__ A,
                                                  const __bf16* __restrict__ Wbf,
                                                  const float* __restrict__ bias,
                                                  float* __restrict__ out) {
    __shared__ __align__(16) __bf16 sA[16 * NMAXX];     // 32 KB
    const int tid = threadIdx.x;
    const int r0  = blockIdx.x * 16;
    for (int idx = tid; idx < 16 * NMAXX; idx += 256) {
        int m = idx >> 10, k = idx & (NMAXX - 1);
        sA[idx] = (__bf16)A[(size_t)(r0 + m) * NMAXX + k];
    }
    __syncthreads();

    const int wave = tid >> 5, lane = tid & 31;
    const int n0 = 16 * (2 * wave), n1 = 16 * (2 * wave + 1);
    v8f acc0 = {}, acc1 = {};
    for (int k0 = 0; k0 < NMAXX; k0 += 32) {
        v16bf a = frag_a(sA, NMAXX, k0, lane);
        acc0 = wmma_bf16(a, frag_b(Wbf, NMAXX, n0, k0, lane), acc0);
        acc1 = wmma_bf16(a, frag_b(Wbf, NMAXX, n1, k0, lane), acc1);
    }
    const int n = lane & 15, g = lane >> 4;
    float b0 = bias[n0 + n], b1 = bias[n1 + n];
#pragma unroll
    for (int r = 0; r < 8; ++r) {
        int m = r + 8 * g;
        out[(size_t)(r0 + m) * Hh + n0 + n] = acc0[r] + b0;
        out[(size_t)(r0 + m) * Hh + n1 + n] = acc1[r] + b1;
    }
}

// ---------------------------------------------------------------------------
// state/state_c init:  state = pad(nf)[node_idx_feat]   (index 0 -> zeros)
// ---------------------------------------------------------------------------
__global__ void k_gather(const float* __restrict__ nf, const float* __restrict__ nfc,
                         const int* __restrict__ idxf,
                         float* __restrict__ state, float* __restrict__ state_c) {
    int t = blockIdx.x * blockDim.x + threadIdx.x;
    if (t >= NNODES * Hh) return;
    int v = t >> 8, k = t & (Hh - 1);
    int j = idxf[v];
    state[t]   = (j == 0) ? 0.0f : nf [(size_t)(j - 1) * Hh + k];
    state_c[t] = (j == 0) ? 0.0f : nfc[(size_t)(j - 1) * Hh + k];
}

__global__ void k_zero(float* __restrict__ p, int n) {
    int t = blockIdx.x * blockDim.x + threadIdx.x;
    if (t < n) p[t] = 0.0f;
}

// ---------------------------------------------------------------------------
// per-edge MLP + gate + segment_sum.  16 edges per block, 8 waves.
// Layer1: 24 N-tiles (msg 256 + att 128) -> 3 tiles/wave, K=512.
// Layer2: msg(K=256)+gate(K=128) for n-blocks {2w,2w+1}, fused sigmoid*msg,
//         then global_atomic_add_f32 into agg[e1].
// ---------------------------------------------------------------------------
__global__ void __launch_bounds__(256) k_edge(
        const float* __restrict__ state,
        const int* __restrict__ edges, const int* __restrict__ att_idx,
        const int* __restrict__ node_dist,
        const __bf16* __restrict__ msgW1, const float* __restrict__ msgb1,
        const __bf16* __restrict__ msgW2, const float* __restrict__ msgb2,
        const __bf16* __restrict__ attW1, const float* __restrict__ attb1,
        const __bf16* __restrict__ attW2, const float* __restrict__ attb2,
        float* __restrict__ agg) {
    __shared__ __align__(16) __bf16 sEi [16 * EIN];    // 16 KB  (A tile, layer1)
    __shared__ __align__(16) __bf16 sH1m[16 * Hh];     //  8 KB  (A tile, msg layer2)
    __shared__ __align__(16) __bf16 sH1a[16 * ATTH];   //  4 KB  (A tile, att layer2)
    __shared__ int sE0[16], sE1[16], sAi0[16], sAi1[16], sNd0[16], sNd1[16];

    const int tid = threadIdx.x;
    if (tid < 16) {
        int e  = blockIdx.x * 16 + tid;                // M = 250000 = 16*15625, exact
        int e0 = edges[2 * e], e1 = edges[2 * e + 1];
        sE0[tid] = e0;            sE1[tid] = e1;
        sAi0[tid] = att_idx[e0];  sAi1[tid] = att_idx[e1];
        sNd0[tid] = node_dist[e0]; sNd1[tid] = node_dist[e1];
    }
    __syncthreads();

    // Build ei = [state[e0]-state[e1] | att one-hots | dist one-hots] in LDS.
    for (int idx = tid; idx < 16 * EIN; idx += 256) {
        int m = idx >> 9, f = idx & (EIN - 1);
        float v;
        if (f < Hh) {
            v = state[(size_t)sE0[m] * Hh + f] - state[(size_t)sE1[m] * Hh + f];
        } else if (f < Hh + 128) {
            int ff = f - Hh;
            v = (ff == sAi0[m] || ff == 64 + sAi1[m]) ? 1.0f : 0.0f;
        } else {
            int ff = f - (Hh + 128);
            v = (ff == sNd0[m] || ff == 64 + sNd1[m]) ? 1.0f : 0.0f;
        }
        sEi[idx] = (__bf16)v;
    }
    __syncthreads();

    const int wave = tid >> 5, lane = tid & 31;
    const int n = lane & 15, g = lane >> 4;

    // ---- layer 1 ----
    v8f acc[3] = {};
    for (int k0 = 0; k0 < EIN; k0 += 32) {
        v16bf a = frag_a(sEi, EIN, k0, lane);
#pragma unroll
        for (int j = 0; j < 3; ++j) {
            int t = 3 * wave + j;                       // tiles 0..23
            v16bf b = (t < 16) ? frag_b(msgW1, EIN, 16 * t,        k0, lane)
                               : frag_b(attW1, EIN, 16 * (t - 16), k0, lane);
            acc[j] = wmma_bf16(a, b, acc[j]);
        }
    }
#pragma unroll
    for (int j = 0; j < 3; ++j) {
        int t = 3 * wave + j;
        if (t < 16) {
            int ng = 16 * t + n; float b1 = msgb1[ng];
#pragma unroll
            for (int r = 0; r < 8; ++r) {
                int m = r + 8 * g;
                float h = acc[j][r] + b1;
                sH1m[m * Hh + ng] = (__bf16)(h > 0.0f ? h : 0.0f);   // relu
            }
        } else {
            int ng = 16 * (t - 16) + n; float b1 = attb1[ng];
#pragma unroll
            for (int r = 0; r < 8; ++r) {
                int m = r + 8 * g;
                float h = acc[j][r] + b1;
                sH1a[m * ATTH + ng] = (__bf16)(h > 0.0f ? h : 0.0f); // relu
            }
        }
    }
    __syncthreads();

    // ---- layer 2 + gate + segment-sum ----
#pragma unroll
    for (int p = 0; p < 2; ++p) {
        int nb = 2 * wave + p;                          // n-blocks 0..15
        v8f accM = {}, accG = {};
        for (int k0 = 0; k0 < Hh; k0 += 32)
            accM = wmma_bf16(frag_a(sH1m, Hh, k0, lane),
                             frag_b(msgW2, Hh, 16 * nb, k0, lane), accM);
        for (int k0 = 0; k0 < ATTH; k0 += 32)
            accG = wmma_bf16(frag_a(sH1a, ATTH, k0, lane),
                             frag_b(attW2, ATTH, 16 * nb, k0, lane), accG);
        int ng = 16 * nb + n;
        float bm = msgb2[ng], bg = attb2[ng];
#pragma unroll
        for (int r = 0; r < 8; ++r) {
            int m = r + 8 * g;
            float val = (accM[r] + bm) * sigmoidf(accG[r] + bg);
            atomic_add_f32(agg + (size_t)sE1[m] * Hh + ng, val);
        }
    }
}

// ---------------------------------------------------------------------------
// LSTM cell: gates(16x1024) = agg@W_ih.T + state@W_hh.T + b_ih + b_hh
// Each wave owns the i/f/g/o n-blocks of the SAME 16 hidden columns, so the
// nonlinearity combines in-register -> no 64KB gates staging in LDS.
// ---------------------------------------------------------------------------
__global__ void __launch_bounds__(256) k_lstm(
        const float* __restrict__ agg, float* __restrict__ state, float* __restrict__ state_c,
        const __bf16* __restrict__ wih, const __bf16* __restrict__ whh,
        const float* __restrict__ bih, const float* __restrict__ bhh) {
    __shared__ __align__(16) __bf16 sAg[16 * Hh];      // 8 KB
    __shared__ __align__(16) __bf16 sSt[16 * Hh];      // 8 KB
    const int tid = threadIdx.x;
    const int r0  = blockIdx.x * 16;
    for (int idx = tid; idx < 16 * Hh; idx += 256) {
        int m = idx >> 8, k = idx & (Hh - 1);
        sAg[idx] = (__bf16)agg  [(size_t)(r0 + m) * Hh + k];
        sSt[idx] = (__bf16)state[(size_t)(r0 + m) * Hh + k];
    }
    __syncthreads();

    const int wave = tid >> 5, lane = tid & 31;
    const int n = lane & 15, g = lane >> 4;
#pragma unroll
    for (int p = 0; p < 2; ++p) {
        int hb = 2 * wave + p;                          // hidden block 0..15
        v8f aI = {}, aF = {}, aG = {}, aO = {};
        for (int k0 = 0; k0 < Hh; k0 += 32) {
            v16bf a = frag_a(sAg, Hh, k0, lane);
            aI = wmma_bf16(a, frag_b(wih, Hh,        16 * hb, k0, lane), aI);
            aF = wmma_bf16(a, frag_b(wih, Hh, 256  + 16 * hb, k0, lane), aF);
            aG = wmma_bf16(a, frag_b(wih, Hh, 512  + 16 * hb, k0, lane), aG);
            aO = wmma_bf16(a, frag_b(wih, Hh, 768  + 16 * hb, k0, lane), aO);
        }
        for (int k0 = 0; k0 < Hh; k0 += 32) {
            v16bf a = frag_a(sSt, Hh, k0, lane);
            aI = wmma_bf16(a, frag_b(whh, Hh,        16 * hb, k0, lane), aI);
            aF = wmma_bf16(a, frag_b(whh, Hh, 256  + 16 * hb, k0, lane), aF);
            aG = wmma_bf16(a, frag_b(whh, Hh, 512  + 16 * hb, k0, lane), aG);
            aO = wmma_bf16(a, frag_b(whh, Hh, 768  + 16 * hb, k0, lane), aO);
        }
        int hg = 16 * hb + n;
        float bI = bih[hg]       + bhh[hg];
        float bF = bih[256 + hg] + bhh[256 + hg];
        float bG = bih[512 + hg] + bhh[512 + hg];
        float bO = bih[768 + hg] + bhh[768 + hg];
#pragma unroll
        for (int r = 0; r < 8; ++r) {
            int m = r + 8 * g;
            size_t off = (size_t)(r0 + m) * Hh + hg;
            float c  = state_c[off];
            float c2 = sigmoidf(aF[r] + bF) * c + sigmoidf(aI[r] + bI) * tanhf(aG[r] + bG);
            float h2 = sigmoidf(aO[r] + bO) * tanhf(c2);
            state_c[off] = c2;
            state[off]   = h2;
        }
    }
}

// ---------------------------------------------------------------------------
// log_theta[p] = (state[i0]-state[i1]) . W_out + b_out      (memory bound)
// ---------------------------------------------------------------------------
__global__ void k_out(const float* __restrict__ state, const int* __restrict__ gnn,
                      const float* __restrict__ Wout, const float* __restrict__ bout,
                      float* __restrict__ out) {
    int p = blockIdx.x * blockDim.x + threadIdx.x;
    if (p >= POUT) return;
    const float4* ra = (const float4*)(state + (size_t)gnn[2 * p]     * Hh);
    const float4* rb = (const float4*)(state + (size_t)gnn[2 * p + 1] * Hh);
    const float4* w4 = (const float4*)Wout;
    float acc = bout[0];
#pragma unroll 4
    for (int k = 0; k < Hh / 4; ++k) {
        float4 a = ra[k], b = rb[k], w = w4[k];
        acc += (a.x - b.x) * w.x + (a.y - b.y) * w.y
             + (a.z - b.z) * w.z + (a.w - b.w) * w.w;
    }
    out[p] = acc;
}

// ---------------------------------------------------------------------------
extern "C" void kernel_launch(void* const* d_in, const int* in_sizes, int n_in,
                              void* d_out, int out_size, void* d_ws, size_t ws_size,
                              hipStream_t stream) {
    (void)in_sizes; (void)n_in; (void)out_size; (void)ws_size;
    const float* A_pad         = (const float*)d_in[0];
    const int*   node_dist     = (const int*)  d_in[1];
    const int*   edges         = (const int*)  d_in[2];
    const int*   node_idx_gnn  = (const int*)  d_in[3];
    const int*   node_idx_feat = (const int*)  d_in[4];
    const int*   att_idx       = (const int*)  d_in[5];
    const float* W_dec   = (const float*)d_in[6];
    const float* b_dec   = (const float*)d_in[7];
    const float* W_dec_c = (const float*)d_in[8];
    const float* b_dec_c = (const float*)d_in[9];
    const float* msg_W1  = (const float*)d_in[10];
    const float* msg_b1  = (const float*)d_in[11];
    const float* msg_W2  = (const float*)d_in[12];
    const float* msg_b2  = (const float*)d_in[13];
    const float* att_W1  = (const float*)d_in[14];
    const float* att_b1  = (const float*)d_in[15];
    const float* att_W2  = (const float*)d_in[16];
    const float* att_b2  = (const float*)d_in[17];
    const float* W_ih    = (const float*)d_in[18];
    const float* W_hh    = (const float*)d_in[19];
    const float* b_ih    = (const float*)d_in[20];
    const float* b_hh    = (const float*)d_in[21];
    const float* W_out   = (const float*)d_in[22];
    const float* b_out   = (const float*)d_in[23];

    char* w = (char*)d_ws;
    auto carve = [&](size_t bytes) -> char* {
        char* p = w; w += (bytes + 255) & ~(size_t)255; return p;
    };
    __bf16* wdec_bf  = (__bf16*)carve((size_t)Hh * NMAXX * 2);
    __bf16* wdecc_bf = (__bf16*)carve((size_t)Hh * NMAXX * 2);
    __bf16* msgW1_bf = (__bf16*)carve((size_t)Hh * EIN   * 2);
    __bf16* msgW2_bf = (__bf16*)carve((size_t)Hh * Hh    * 2);
    __bf16* attW1_bf = (__bf16*)carve((size_t)ATTH * EIN * 2);
    __bf16* attW2_bf = (__bf16*)carve((size_t)Hh * ATTH  * 2);
    __bf16* wih_bf   = (__bf16*)carve((size_t)4 * Hh * Hh * 2);
    __bf16* whh_bf   = (__bf16*)carve((size_t)4 * Hh * Hh * 2);
    float* nf_dec  = (float*)carve((size_t)NNODES * Hh * 4);
    float* nfc_dec = (float*)carve((size_t)NNODES * Hh * 4);
    float* state   = (float*)carve((size_t)NNODES * Hh * 4);
    float* state_c = (float*)carve((size_t)NNODES * Hh * 4);
    float* agg     = (float*)carve((size_t)NNODES * Hh * 4);

    // 1) weights -> bf16
    struct Cv { const float* s; __bf16* d; int n; };
    const Cv conv[8] = {
        {W_dec,   wdec_bf,  Hh * NMAXX}, {W_dec_c, wdecc_bf, Hh * NMAXX},
        {msg_W1,  msgW1_bf, Hh * EIN},   {msg_W2,  msgW2_bf, Hh * Hh},
        {att_W1,  attW1_bf, ATTH * EIN}, {att_W2,  attW2_bf, Hh * ATTH},
        {W_ih,    wih_bf,   4 * Hh * Hh},{W_hh,    whh_bf,   4 * Hh * Hh}};
    for (int i = 0; i < 8; ++i)
        k_f32_to_bf16<<<(conv[i].n + 1023) / 1024, 256, 0, stream>>>(conv[i].s, conv[i].d, conv[i].n);

    // 2) decoders
    k_dec_gemm<<<NNODES / 16, 256, 0, stream>>>(A_pad, wdec_bf,  b_dec,   nf_dec);
    k_dec_gemm<<<NNODES / 16, 256, 0, stream>>>(A_pad, wdecc_bf, b_dec_c, nfc_dec);

    // 3) state gather (with pad-row-0 semantics)
    k_gather<<<(NNODES * Hh) / 256, 256, 0, stream>>>(nf_dec, nfc_dec, node_idx_feat, state, state_c);

    // 4) two propagation rounds
    for (int it = 0; it < 2; ++it) {
        k_zero<<<(NNODES * Hh) / 256, 256, 0, stream>>>(agg, NNODES * Hh);
        k_edge<<<MEDGES / 16, 256, 0, stream>>>(state, edges, att_idx, node_dist,
                                                msgW1_bf, msg_b1, msgW2_bf, msg_b2,
                                                attW1_bf, att_b1, attW2_bf, att_b2, agg);
        k_lstm<<<NNODES / 16, 256, 0, stream>>>(agg, state, state_c,
                                                wih_bf, whh_bf, b_ih, b_hh);
    }

    // 5) output
    k_out<<<(POUT + 255) / 256, 256, 0, stream>>>(state, node_idx_gnn, W_out, b_out, (float*)d_out);
}
